// RaFlow_VoD_79706003079890
// MI455X (gfx1250) — matrix operand order repaired and checked by reading, not compiled, not run on hardware
//
#include <hip/hip_runtime.h>
#include <hip/hip_bf16.h>
#include <cstdint>
#include <cstddef>

typedef __bf16 bf16;
typedef __attribute__((ext_vector_type(16))) __bf16 v16bf;
typedef __attribute__((ext_vector_type(8)))  __bf16 v8bf;
typedef __attribute__((ext_vector_type(8)))  float  v8f;
typedef int v4i __attribute__((vector_size(16)));

// ---------------------------------------------------------------------------
// CDNA5 async global->LDS path (ASYNCcnt), with synchronous fallback.
// Builtin signature (from probe): (v4i AS1* gsrc, v4i AS3* ldst, Imm off, Imm cpol)
// ---------------------------------------------------------------------------
#if defined(__HIP_DEVICE_COMPILE__) && defined(__gfx1250__) && \
    __has_builtin(__builtin_amdgcn_global_load_async_to_lds_b128)
#define HAS_ASYNC 1
#else
#define HAS_ASYNC 0
#endif

#define GPTR(p) ((__attribute__((address_space(1))) v4i*)(void*)(p))
#define LPTR(p) ((__attribute__((address_space(3))) v4i*)(void*)(p))

#if HAS_ASYNC
#if __has_builtin(__builtin_amdgcn_s_wait_asynccnt)
#define WAIT_ASYNC(n) __builtin_amdgcn_s_wait_asynccnt(n)
#else
#define WAIT_ASYNC(n) asm volatile("s_wait_asynccnt %0" ::"i"(n) : "memory")
#endif
#else
#define WAIT_ASYNC(n)
#endif

// ---------------------------------------------------------------------------
// Generic helpers
// ---------------------------------------------------------------------------
static inline int ceilDiv(int a, int b) { return (a + b - 1) / b; }

__global__ void transpose3_kernel(const float* __restrict__ src, // (B,3,N)
                                  float* __restrict__ dst,       // (B,N,3)
                                  int N) {
  int i = blockIdx.x * blockDim.x + threadIdx.x;
  int total = 2 * N * 3;
  if (i >= total) return;
  int b = i / (N * 3);
  int rem = i % (N * 3);
  int n = rem / 3, c = rem % 3;
  dst[i] = src[((size_t)b * 3 + c) * N + n];
}

__global__ void convert_w_kernel(const float* __restrict__ src, // (n,k)
                                 bf16* __restrict__ dst,        // (n,kp)
                                 int n, int k, int kp) {
  int i = blockIdx.x * blockDim.x + threadIdx.x;
  if (i >= n * kp) return;
  int r = i / kp, c = i % kp;
  float v = (c < k) ? src[(size_t)r * k + c] : 0.f;
  dst[i] = (bf16)v;
}

__global__ void f32_to_bf16_kernel(const float* __restrict__ s, bf16* __restrict__ d, int n) {
  int i = blockIdx.x * blockDim.x + threadIdx.x;
  if (i < n) d[i] = (bf16)s[i];
}

__global__ void colmax_kernel(const float* __restrict__ x, // (B,N,C)
                              float* __restrict__ out,     // (B,C)
                              int N, int C) {
  int i = blockIdx.x * blockDim.x + threadIdx.x;
  if (i >= 2 * C) return;
  int b = i / C, c = i % C;
  float m = -1e30f;
  for (int n = 0; n < N; ++n) m = fmaxf(m, x[((size_t)b * N + n) * C + c]);
  out[i] = m;
}

// ---------------------------------------------------------------------------
// KNN: per query point keep sorted (ascending dist) top-K of target cloud.
// ---------------------------------------------------------------------------
__global__ __launch_bounds__(256) void knn_kernel(const float* __restrict__ q, // (B,N,3)
                                                  const float* __restrict__ t, // (B,N,3)
                                                  int* __restrict__ idxOut,    // (B,N,K)
                                                  int K, int N) {
  int b = blockIdx.y;
  int i = blockIdx.x * blockDim.x + threadIdx.x;
  __shared__ float ts[1024 * 3];
  for (int j = threadIdx.x; j < N * 3; j += blockDim.x) ts[j] = t[(size_t)b * N * 3 + j];
  __syncthreads();
  if (i >= N) return;
  float qx = q[((size_t)b * N + i) * 3 + 0];
  float qy = q[((size_t)b * N + i) * 3 + 1];
  float qz = q[((size_t)b * N + i) * 3 + 2];
  float bestd[32];
  int besti[32];
  for (int s = 0; s < K; ++s) { bestd[s] = 1e30f; besti[s] = 0; }
  for (int j = 0; j < N; ++j) {
    float dx = ts[j * 3 + 0] - qx;
    float dy = ts[j * 3 + 1] - qy;
    float dz = ts[j * 3 + 2] - qz;
    float d = dx * dx + dy * dy + dz * dz;
    if (d < bestd[K - 1]) {
      int p = K - 1;
      while (p > 0 && d < bestd[p - 1]) { bestd[p] = bestd[p - 1]; besti[p] = besti[p - 1]; --p; }
      bestd[p] = d; besti[p] = j;
    }
  }
  for (int s = 0; s < K; ++s) idxOut[((size_t)b * N + i) * K + s] = besti[s];
}

// ---------------------------------------------------------------------------
// WMMA bf16 GEMM:  Y[M,N] = act(X[M,K] @ W[N,K]^T + bias[N])
// BM=128, BN=64, BK=32; 8 waves, each wave 2x2 16x16 tiles.
// Double-buffered LDS; async global->LDS DMA when available (ASYNCcnt).
// act: 0 none, 1 relu, 2 leaky-relu(0.1)
// ---------------------------------------------------------------------------
#define BM 128
#define BN 64
#define BK 32

__global__ __launch_bounds__(256) void gemm_bf16_kernel(
    const bf16* __restrict__ X, const bf16* __restrict__ W, const float* __restrict__ bias,
    float* __restrict__ Yf, bf16* __restrict__ Yb, int M, int N, int K, int ldf, int co, int act) {
  __shared__ __attribute__((aligned(64))) bf16 As[2][BM * BK];
  __shared__ __attribute__((aligned(64))) bf16 Bs[2][BN * BK];
  const int tid = threadIdx.x;
  const int wave = tid >> 5, lane = tid & 31;
  const int wm = wave & 3, wn = wave >> 2;
  const int h = lane >> 4, l16 = lane & 15;
  const int bm = blockIdx.x * BM, bn = blockIdx.y * BN;

  // Per-thread staging coordinates: 16-byte (8 x bf16) chunks.
  // A tile: BM*BK/8 = 512 chunks -> 2 per thread.  B tile: 256 chunks -> 1.
  const int ar0 = tid >> 2;                 // rows 0..63
  const int ar1 = (tid + 256) >> 2;         // rows 64..127
  const int ac = (tid & 3) << 3;            // k-offset within BK

  auto stage = [&](int buf, int kk) {
    const bf16* g0 = &X[(size_t)(bm + ar0) * K + kk + ac];
    const bf16* g1 = &X[(size_t)(bm + ar1) * K + kk + ac];
    const bf16* g2 = &W[(size_t)(bn + ar0) * K + kk + ac];
    bf16* l0 = &As[buf][ar0 * BK + ac];
    bf16* l1 = &As[buf][ar1 * BK + ac];
    bf16* l2 = &Bs[buf][ar0 * BK + ac];
#if HAS_ASYNC
    __builtin_amdgcn_global_load_async_to_lds_b128(GPTR(g0), LPTR(l0), 0, 0);
    __builtin_amdgcn_global_load_async_to_lds_b128(GPTR(g1), LPTR(l1), 0, 0);
    __builtin_amdgcn_global_load_async_to_lds_b128(GPTR(g2), LPTR(l2), 0, 0);
#else
    *(v8bf*)l0 = *(const v8bf*)g0;
    *(v8bf*)l1 = *(const v8bf*)g1;
    *(v8bf*)l2 = *(const v8bf*)g2;
#endif
  };

  const int nk = K / BK;
  v8f acc[2][2] = {};
  stage(0, 0);
  for (int t = 0; t < nk; ++t) {
    const int cur = t & 1;
    if (t + 1 < nk) stage(cur ^ 1, (t + 1) * BK);
#if HAS_ASYNC
    if (t + 1 < nk) { WAIT_ASYNC(3); } else { WAIT_ASYNC(0); }
#endif
    __syncthreads();
    v16bf a[2], bfr[2];
#pragma unroll
    for (int i = 0; i < 2; ++i) {
      const bf16* ap = &As[cur][(wm * 32 + i * 16 + l16) * BK];
      a[i] = *(const v16bf*)(ap + h * 16);
      const bf16* bp = &Bs[cur][(wn * 32 + i * 16 + l16) * BK];
      bfr[i] = *(const v16bf*)(bp + h * 16);
    }
#pragma unroll
    for (int i = 0; i < 2; ++i)
#pragma unroll
      for (int j = 0; j < 2; ++j)
        acc[i][j] = __builtin_amdgcn_wmma_f32_16x16x32_bf16(false, a[i], false, bfr[j],
                                                            (short)0, acc[i][j], false, false);
    __syncthreads();
  }
#pragma unroll
  for (int i = 0; i < 2; ++i)
#pragma unroll
    for (int j = 0; j < 2; ++j) {
      int n = bn + wn * 32 + j * 16 + l16;
      float bv = bias ? bias[n] : 0.f;
#pragma unroll
      for (int r = 0; r < 8; ++r) {
        int m = bm + wm * 32 + i * 16 + h * 8 + r;
        float v = acc[i][j][r] + bv;
        if (act == 1) v = fmaxf(v, 0.f);
        else if (act == 2) v = v > 0.f ? v : 0.1f * v;
        if (Yf) Yf[(size_t)m * ldf + co + n] = v;
        if (Yb) Yb[(size_t)m * N + n] = (bf16)v;
      }
    }
}

// ---------------------------------------------------------------------------
// mse block: per (b,n,j) 6->32->32->64 relu MLP, atomic max over neighbors.
// ---------------------------------------------------------------------------
__global__ void mse_block_kernel(const float* __restrict__ xt, const float* __restrict__ ft,
                                 const int* __restrict__ kn, int k,
                                 const float* __restrict__ W1, const float* __restrict__ b1,
                                 const float* __restrict__ W2, const float* __restrict__ b2,
                                 const float* __restrict__ W3, const float* __restrict__ b3,
                                 unsigned int* __restrict__ pre, int N) {
  int gid = blockIdx.x * blockDim.x + threadIdx.x;
  if (gid >= 2 * N * k) return;
  int j = gid % k;
  int pg = gid / k;
  int b = pg / N;
  int nb = kn[(size_t)pg * 32 + j];
  int nbg = b * N + nb;
  float in6[6];
  in6[0] = ft[(size_t)nbg * 3 + 0];
  in6[1] = ft[(size_t)nbg * 3 + 1];
  in6[2] = ft[(size_t)nbg * 3 + 2];
  in6[3] = xt[(size_t)nbg * 3 + 0] - xt[(size_t)pg * 3 + 0];
  in6[4] = xt[(size_t)nbg * 3 + 1] - xt[(size_t)pg * 3 + 1];
  in6[5] = xt[(size_t)nbg * 3 + 2] - xt[(size_t)pg * 3 + 2];
  float h1[32];
  for (int c = 0; c < 32; ++c) {
    float s = b1[c];
#pragma unroll
    for (int d = 0; d < 6; ++d) s += W1[c * 6 + d] * in6[d];
    h1[c] = fmaxf(s, 0.f);
  }
  float h2[32];
  for (int c = 0; c < 32; ++c) {
    float s = b2[c];
    for (int d = 0; d < 32; ++d) s += W2[c * 32 + d] * h1[d];
    h2[c] = fmaxf(s, 0.f);
  }
  for (int c = 0; c < 64; ++c) {
    float s = b3[c];
    for (int d = 0; d < 32; ++d) s += W3[c * 32 + d] * h2[d];
    s = fmaxf(s, 0.f);
    atomicMax(&pre[(size_t)pg * 64 + c], __float_as_uint(s));
  }
}

// ---------------------------------------------------------------------------
// fc gather: rows (b,n,j) -> [pf1full(512), pf2full(512), dir(3), pad] bf16
// ---------------------------------------------------------------------------
__global__ void gather_fc_kernel(const float* __restrict__ pf1, const float* __restrict__ pf1m,
                                 const float* __restrict__ pf2, const float* __restrict__ pf2m,
                                 const float* __restrict__ x1, const float* __restrict__ x2,
                                 const int* __restrict__ knnc, bf16* __restrict__ X, int N,
                                 int rowBase, int rows) {
  int gid = blockIdx.x * blockDim.x + threadIdx.x;
  if (gid >= rows * 1056) return;
  int r = gid / 1056, c = gid % 1056;
  int s = rowBase + r;
  int j = s % 8;
  int pg = s / 8;
  int b = pg / N;
  int nb = knnc[(size_t)pg * 8 + j];
  int nbg = b * N + nb;
  float v;
  if (c < 512) {
    v = (c < 256) ? pf1[(size_t)pg * 256 + c] : pf1m[b * 256 + (c - 256)];
  } else if (c < 1024) {
    int cc = c - 512;
    v = (cc < 256) ? pf2[(size_t)nbg * 256 + cc] : pf2m[b * 256 + (cc - 256)];
  } else if (c < 1027) {
    int d = c - 1024;
    v = x2[(size_t)nbg * 3 + d] - x1[(size_t)pg * 3 + d];
  } else {
    v = 0.f;
  }
  X[(size_t)r * 1056 + c] = (bf16)v;
}

// weight net 3->8->8->512 (relu), per neighbor row
__global__ void wn_kernel(const float* __restrict__ xq, const float* __restrict__ xt,
                          const int* __restrict__ idx, int idxStride,
                          const float* __restrict__ W1, const float* __restrict__ b1,
                          const float* __restrict__ W2, const float* __restrict__ b2,
                          const float* __restrict__ W3, const float* __restrict__ b3,
                          float* __restrict__ wOut, int N, int rowBase, int rows) {
  int r = blockIdx.x * blockDim.x + threadIdx.x;
  if (r >= rows) return;
  int s = rowBase + r;
  int j = s % 8;
  int pg = s / 8;
  int b = pg / N;
  int nb = idx[(size_t)pg * idxStride + j];
  int nbg = b * N + nb;
  float d0 = xt[(size_t)nbg * 3 + 0] - xq[(size_t)pg * 3 + 0];
  float d1 = xt[(size_t)nbg * 3 + 1] - xq[(size_t)pg * 3 + 1];
  float d2 = xt[(size_t)nbg * 3 + 2] - xq[(size_t)pg * 3 + 2];
  float h1[8], h2[8];
#pragma unroll
  for (int c = 0; c < 8; ++c)
    h1[c] = fmaxf(b1[c] + W1[c * 3 + 0] * d0 + W1[c * 3 + 1] * d1 + W1[c * 3 + 2] * d2, 0.f);
#pragma unroll
  for (int c = 0; c < 8; ++c) {
    float s2 = b2[c];
#pragma unroll
    for (int d = 0; d < 8; ++d) s2 += W2[c * 8 + d] * h1[d];
    h2[c] = fmaxf(s2, 0.f);
  }
  for (int c = 0; c < 512; ++c) {
    float s3 = b3[c];
#pragma unroll
    for (int d = 0; d < 8; ++d) s3 += W3[c * 8 + d] * h2[d];
    wOut[(size_t)r * 512 + c] = fmaxf(s3, 0.f);
  }
}

__global__ void p2p_sum_kernel(const float* __restrict__ w, const float* __restrict__ h,
                               float* __restrict__ p2p, int pBase, int pts) {
  int i = blockIdx.x * blockDim.x + threadIdx.x;
  if (i >= pts * 512) return;
  int pl = i / 512, c = i % 512;
  float s = 0.f;
  for (int j = 0; j < 8; ++j) {
    size_t r = (size_t)(pl * 8 + j) * 512 + c;
    s += w[r] * h[r];
  }
  p2p[((size_t)(pBase + pl)) * 512 + c] = s;
}

__global__ void cor_kernel(const float* __restrict__ w2, const float* __restrict__ p2p,
                           const int* __restrict__ knn32, float* __restrict__ cor, int N,
                           int pBase, int pts) {
  int i = blockIdx.x * blockDim.x + threadIdx.x;
  if (i >= pts * 512) return;
  int pl = i / 512, c = i % 512;
  int pg = pBase + pl;
  int b = pg / N;
  float s = 0.f;
  for (int j = 0; j < 8; ++j) {
    int nb = knn32[(size_t)pg * 32 + j];
    s += w2[(size_t)(pl * 8 + j) * 512 + c] * p2p[((size_t)(b * N + nb)) * 512 + c];
  }
  cor[(size_t)pg * 512 + c] = s;
}

__global__ void emb_kernel(const float* __restrict__ f1t, const float* __restrict__ pf1,
                           const float* __restrict__ pf1m, const float* __restrict__ cor,
                           float* __restrict__ emb, int N) {
  int i = blockIdx.x * blockDim.x + threadIdx.x;
  if (i >= 2 * N * 1027) return;
  int pg = i / 1027, c = i % 1027;
  int b = pg / N;
  float v;
  if (c < 3) v = f1t[(size_t)pg * 3 + c];
  else if (c < 515) {
    int cc = c - 3;
    v = (cc < 256) ? pf1[(size_t)pg * 256 + cc] : pf1m[b * 256 + (cc - 256)];
  } else v = cor[(size_t)pg * 512 + (c - 515)];
  emb[(size_t)pg * 1027 + c] = v;
}

__global__ void gather_fd_kernel(const float* __restrict__ emb, const float* __restrict__ x1,
                                 const int* __restrict__ knn32, int k, bf16* __restrict__ X,
                                 int N, int rowBase, int rows) {
  int gid = blockIdx.x * blockDim.x + threadIdx.x;
  if (gid >= rows * 1056) return;
  int r = gid / 1056, c = gid % 1056;
  int s = rowBase + r;
  int j = s % k;
  int pg = s / k;
  int b = pg / N;
  int nb = knn32[(size_t)pg * 32 + j];
  int nbg = b * N + nb;
  float v;
  if (c < 1027) v = emb[(size_t)nbg * 1027 + c];
  else if (c < 1030) {
    int d = c - 1027;
    v = x1[(size_t)nbg * 3 + d] - x1[(size_t)pg * 3 + d];
  } else v = 0.f;
  X[(size_t)r * 1056 + c] = (bf16)v;
}

__global__ void fdmax_kernel(const float* __restrict__ Y3, unsigned int* __restrict__ pre, int k,
                             int rowBase, int rows) {
  int pts = rows / k;
  int i = blockIdx.x * blockDim.x + threadIdx.x;
  if (i >= pts * 64) return;
  int pl = i / 64, c = i % 64;
  int pg = (rowBase + pl * k) / k;
  float m = 0.f;
  for (int j = 0; j < k; ++j) m = fmaxf(m, Y3[(size_t)(pl * k + j) * 64 + c]);
  atomicMax(&pre[(size_t)pg * 64 + c], __float_as_uint(m));
}

__global__ void gather_fp_kernel(const float* __restrict__ prop, const float* __restrict__ propm,
                                 bf16* __restrict__ X, int N) {
  int i = blockIdx.x * blockDim.x + threadIdx.x;
  if (i >= 2 * N * 512) return;
  int pg = i / 512, c = i % 512;
  int b = pg / N;
  float v = (c < 256) ? prop[(size_t)pg * 256 + c] : propm[b * 256 + (c - 256)];
  X[i] = (bf16)v;
}

__global__ void fp_out_kernel(const float* __restrict__ Y, const float* __restrict__ W,
                              const float* __restrict__ bb, float* __restrict__ outBCN, int N) {
  int pg = blockIdx.x * blockDim.x + threadIdx.x;
  if (pg >= 2 * N) return;
  int b = pg / N, n = pg % N;
  for (int c = 0; c < 3; ++c) {
    float s = bb[c];
    for (int d = 0; d < 64; ++d) s += W[c * 64 + d] * Y[(size_t)pg * 64 + d];
    outBCN[((size_t)b * 3 + c) * N + n] = s;
  }
}

// ---------------------------------------------------------------------------
// Rigid fit (Kabsch) + 3x3 SVD
// ---------------------------------------------------------------------------
__device__ inline float block_sum256(float v, float* red) {
  int tid = threadIdx.x;
  red[tid] = v;
  __syncthreads();
  for (int s = 128; s > 0; s >>= 1) {
    if (tid < s) red[tid] += red[tid + s];
    __syncthreads();
  }
  float r = red[0];
  __syncthreads();
  return r;
}

__device__ void svd_rot3(const float Hm[9], float R[9]) {
  float S[9], V[9] = {1, 0, 0, 0, 1, 0, 0, 0, 1};
  for (int i = 0; i < 3; ++i)
    for (int j = 0; j < 3; ++j) {
      float s = 0;
      for (int kq = 0; kq < 3; ++kq) s += Hm[kq * 3 + i] * Hm[kq * 3 + j];
      S[i * 3 + j] = s;
    }
  const int pp[3] = {0, 0, 1}, qq[3] = {1, 2, 2};
  for (int sweep = 0; sweep < 12; ++sweep) {
    int p = pp[sweep % 3], q = qq[sweep % 3];
    float apq = S[p * 3 + q];
    if (fabsf(apq) < 1e-20f) continue;
    float app = S[p * 3 + p], aqq = S[q * 3 + q];
    float tau = (aqq - app) / (2.0f * apq);
    float t = (tau >= 0.f ? 1.f : -1.f) / (fabsf(tau) + sqrtf(1.f + tau * tau));
    float c = 1.f / sqrtf(1.f + t * t), s = t * c;
    for (int kq = 0; kq < 3; ++kq) {
      float skp = S[kq * 3 + p], skq = S[kq * 3 + q];
      S[kq * 3 + p] = c * skp - s * skq;
      S[kq * 3 + q] = s * skp + c * skq;
    }
    for (int kq = 0; kq < 3; ++kq) {
      float spk = S[p * 3 + kq], sqk = S[q * 3 + kq];
      S[p * 3 + kq] = c * spk - s * sqk;
      S[q * 3 + kq] = s * spk + c * sqk;
    }
    for (int kq = 0; kq < 3; ++kq) {
      float vkp = V[kq * 3 + p], vkq = V[kq * 3 + q];
      V[kq * 3 + p] = c * vkp - s * vkq;
      V[kq * 3 + q] = s * vkp + c * vkq;
    }
  }
  float U[9], sig[3];
  for (int i = 0; i < 3; ++i) {
    float u0 = 0, u1 = 0, u2 = 0;
    for (int kq = 0; kq < 3; ++kq) {
      u0 += Hm[0 * 3 + kq] * V[kq * 3 + i];
      u1 += Hm[1 * 3 + kq] * V[kq * 3 + i];
      u2 += Hm[2 * 3 + kq] * V[kq * 3 + i];
    }
    float sg = sqrtf(fmaxf(S[i * 3 + i], 0.f));
    sig[i] = sg;
    float inv = sg > 1e-12f ? 1.f / sg : 0.f;
    U[0 * 3 + i] = u0 * inv;
    U[1 * 3 + i] = u1 * inv;
    U[2 * 3 + i] = u2 * inv;
  }
  for (int i = 0; i < 3; ++i)
    if (sig[i] <= 1e-12f) {
      int a = (i + 1) % 3, b = (i + 2) % 3;
      float cx = U[1 * 3 + a] * U[2 * 3 + b] - U[2 * 3 + a] * U[1 * 3 + b];
      float cy = U[2 * 3 + a] * U[0 * 3 + b] - U[0 * 3 + a] * U[2 * 3 + b];
      float cz = U[0 * 3 + a] * U[1 * 3 + b] - U[1 * 3 + a] * U[0 * 3 + b];
      float nn = sqrtf(cx * cx + cy * cy + cz * cz);
      if (nn < 1e-20f) nn = 1.f;
      U[0 * 3 + i] = cx / nn; U[1 * 3 + i] = cy / nn; U[2 * 3 + i] = cz / nn;
    }
  float Mm[9];
  for (int i = 0; i < 3; ++i)
    for (int j = 0; j < 3; ++j) {
      float s = 0;
      for (int kq = 0; kq < 3; ++kq) s += V[i * 3 + kq] * U[j * 3 + kq];
      Mm[i * 3 + j] = s;
    }
  float det = Mm[0] * (Mm[4] * Mm[8] - Mm[5] * Mm[7]) - Mm[1] * (Mm[3] * Mm[8] - Mm[5] * Mm[6]) +
              Mm[2] * (Mm[3] * Mm[7] - Mm[4] * Mm[6]);
  if (det < 0.f)
    for (int j = 0; j < 3; ++j) Mm[2 * 3 + j] = -Mm[2 * 3 + j];
  for (int i = 0; i < 9; ++i) R[i] = Mm[i];
}

__global__ __launch_bounds__(256) void rigid_kernel(const float* __restrict__ pc1,
                                                    const float* __restrict__ flow,
                                                    const float* __restrict__ wts,
                                                    float* __restrict__ trans, int N) {
  __shared__ float red[256];
  int b = blockIdx.x, tid = threadIdx.x;
  float sA[3] = {0, 0, 0}, sB[3] = {0, 0, 0};
  for (int n = tid; n < N; n += 256) {
    float w = wts ? wts[b * N + n] : 1.f;
#pragma unroll
    for (int c = 0; c < 3; ++c) {
      float a = pc1[((size_t)b * 3 + c) * N + n];
      float wp = a + flow[((size_t)b * 3 + c) * N + n];
      sA[c] += w * a;
      sB[c] += w * wp;
    }
  }
  float cA[3], cB[3];
  for (int c = 0; c < 3; ++c) {
    cA[c] = block_sum256(sA[c], red) / N;
    cB[c] = block_sum256(sB[c], red) / N;
  }
  float sH[9] = {};
  for (int n = tid; n < N; n += 256) {
    float w = wts ? wts[b * N + n] : 1.f;
    float am[3], bm[3];
#pragma unroll
    for (int c = 0; c < 3; ++c) {
      float a = pc1[((size_t)b * 3 + c) * N + n];
      am[c] = a - cA[c];
      bm[c] = a + flow[((size_t)b * 3 + c) * N + n] - cB[c];
    }
#pragma unroll
    for (int i = 0; i < 3; ++i)
#pragma unroll
      for (int j = 0; j < 3; ++j) sH[i * 3 + j] += am[i] * bm[j] * w;
  }
  float H[9];
  for (int i = 0; i < 9; ++i) H[i] = block_sum256(sH[i], red);
  if (tid == 0) {
    float R[9];
    svd_rot3(H, R);
    float* T = trans + b * 16;
    for (int r = 0; r < 3; ++r) {
      float t = -(R[r * 3 + 0] * cA[0] + R[r * 3 + 1] * cA[1] + R[r * 3 + 2] * cA[2]) + cB[r];
      T[r * 4 + 0] = R[r * 3 + 0];
      T[r * 4 + 1] = R[r * 3 + 1];
      T[r * 4 + 2] = R[r * 3 + 2];
      T[r * 4 + 3] = t;
    }
    T[12] = 0.f; T[13] = 0.f; T[14] = 0.f; T[15] = 1.f;
  }
}

__global__ void mask_kernel(const float* __restrict__ pc1, const float* __restrict__ feature1,
                            const float* __restrict__ interval, const float* __restrict__ trans1,
                            float* __restrict__ maskf, float* __restrict__ countf, int N) {
  int i = blockIdx.x * blockDim.x + threadIdx.x;
  if (i >= 2 * N) return;
  int b = i / N, n = i % N;
  const float* T = trans1 + b * 16;
  float p[3];
#pragma unroll
  for (int c = 0; c < 3; ++c) p[c] = pc1[((size_t)b * 3 + c) * N + n];
  float sf[3];
#pragma unroll
  for (int r = 0; r < 3; ++r)
    sf[r] = T[r * 4 + 0] * p[0] + T[r * 4 + 1] * p[1] + T[r * 4 + 2] * p[2] + T[r * 4 + 3] - p[r];
  float pn = sqrtf(p[0] * p[0] + p[1] * p[1] + p[2] * p[2]);
  float proj = (sf[0] * p[0] + sf[1] * p[1] + sf[2] * p[2]) / pn;
  float vel = feature1[((size_t)b * 3 + 0) * N + n];
  float res = vel * interval[b] - proj;
  float m = (fabsf(res / vel) < 0.15f) ? 1.f : 0.f;
  maskf[i] = m;
  atomicAdd(&countf[b], m);
}

__global__ void final_kernel(const float* __restrict__ pc1, const float* __restrict__ outp,
                             const float* __restrict__ t1, const float* __restrict__ t2,
                             const float* __restrict__ maskf, const float* __restrict__ countf,
                             float* __restrict__ sf_agg, float* __restrict__ pre_trans, int N) {
  int i = blockIdx.x * blockDim.x + threadIdx.x;
  if (i >= 2 * N) return;
  int b = i / N, n = i % N;
  bool cond = countf[b] / (float)N > 0.25f;
  float p[3];
#pragma unroll
  for (int c = 0; c < 3; ++c) p[c] = pc1[((size_t)b * 3 + c) * N + n];
  const float* T2 = t2 + b * 16;
  float sf2[3];
#pragma unroll
  for (int r = 0; r < 3; ++r)
    sf2[r] = T2[r * 4 + 0] * p[0] + T2[r * 4 + 1] * p[1] + T2[r * 4 + 2] * p[2] + T2[r * 4 + 3] - p[r];
  float m = maskf[b * N + n];
#pragma unroll
  for (int c = 0; c < 3; ++c) {
    float o = outp[((size_t)b * 3 + c) * N + n];
    sf_agg[((size_t)b * 3 + c) * N + n] = (cond && m > 0.5f) ? sf2[c] : o;
  }
  if (n < 16) {
    const float* Ts = (cond ? t2 : t1) + b * 16;
    pre_trans[b * 16 + n] = Ts[n];
  }
}

// ---------------------------------------------------------------------------
// Host orchestration
// ---------------------------------------------------------------------------
extern "C" void kernel_launch(void* const* d_in, const int* in_sizes, int n_in, void* d_out,
                              int out_size, void* d_ws, size_t ws_size, hipStream_t stream) {
  (void)in_sizes; (void)out_size; (void)ws_size;
  const int B = 2, N = 1024;
  const int SA[4] = {4, 8, 16, 32};
  const int IDX_FC = 5, IDX_WN1 = 11, IDX_WN2 = 17, IDX_FD = 23, IDX_FP = 71, IDX_MSE = 79;

  const float* P[160];
  for (int i = 0; i < n_in && i < 160; ++i) P[i] = (const float*)d_in[i];

  float* outp = (float*)d_out;            // (B,3,N)
  float* sf_agg = outp + B * 3 * N;       // (B,3,N)
  float* pre_trans = sf_agg + B * 3 * N;  // (B,4,4)
  float* mask_out = pre_trans + B * 16;   // (B,N) as 0/1 float

  size_t off = 0;
  auto alloc = [&](size_t bytes) -> void* {
    void* p = (void*)((char*)d_ws + off);
    off += (bytes + 255) & ~(size_t)255;
    return p;
  };

  float* x1 = (float*)alloc((size_t)B * N * 3 * 4);
  float* x2 = (float*)alloc((size_t)B * N * 3 * 4);
  float* f1t = (float*)alloc((size_t)B * N * 3 * 4);
  float* f2t = (float*)alloc((size_t)B * N * 3 * 4);
  int* knn1 = (int*)alloc((size_t)B * N * 32 * 4);
  int* knn2 = (int*)alloc((size_t)B * N * 32 * 4);
  int* knnc = (int*)alloc((size_t)B * N * 8 * 4);
  float* pf1 = (float*)alloc((size_t)B * N * 256 * 4);
  float* pf2 = (float*)alloc((size_t)B * N * 256 * 4);
  float* pf1m = (float*)alloc((size_t)B * 256 * 4);
  float* pf2m = (float*)alloc((size_t)B * 256 * 4);
  float* pre64 = (float*)alloc((size_t)B * N * 64 * 4);
  bf16* Xm64 = (bf16*)alloc((size_t)B * N * 64 * 2);
  bf16* Ym64a = (bf16*)alloc((size_t)B * N * 64 * 2);
  bf16* Ym64b = (bf16*)alloc((size_t)B * N * 64 * 2);
  float* prop = (float*)alloc((size_t)B * N * 256 * 4);
  float* propm = (float*)alloc((size_t)B * 256 * 4);
  float* p2p = (float*)alloc((size_t)B * N * 512 * 4);
  float* cor = (float*)alloc((size_t)B * N * 512 * 4);
  float* emb = (float*)alloc((size_t)B * N * 1027 * 4);
  bf16* Xbig = (bf16*)alloc((size_t)8192 * 1056 * 2);
  bf16* Yba = (bf16*)alloc((size_t)8192 * 512 * 2);
  bf16* Ybb = (bf16*)alloc((size_t)8192 * 512 * 2);
  float* hF32 = (float*)alloc((size_t)8192 * 512 * 4);
  float* wF32 = (float*)alloc((size_t)8192 * 512 * 4);
  float* Y3f = (float*)alloc((size_t)8192 * 64 * 4);
  bf16* Xfp = (bf16*)alloc((size_t)B * N * 512 * 2);
  bf16* fpYa = (bf16*)alloc((size_t)B * N * 256 * 2);
  bf16* fpYb = (bf16*)alloc((size_t)B * N * 128 * 2);
  float* fpY3 = (float*)alloc((size_t)B * N * 64 * 4);
  float* trans1 = (float*)alloc(B * 16 * 4);
  float* trans2 = (float*)alloc(B * 16 * 4);
  float* countf = (float*)alloc(B * 4);

  auto convW = [&](int idx, int n, int k, int kp) -> bf16* {
    bf16* dst = (bf16*)alloc((size_t)n * kp * 2);
    int tot = n * kp;
    convert_w_kernel<<<ceilDiv(tot, 256), 256, 0, stream>>>(P[idx], dst, n, k, kp);
    return dst;
  };
  auto gemm = [&](const bf16* X, const bf16* W, const float* bias, float* Yf, int ldf, int co,
                  bf16* Yb, int M, int Nn, int K, int act) {
    dim3 g(M / BM, Nn / BN);
    gemm_bf16_kernel<<<g, 256, 0, stream>>>(X, W, bias, Yf, Yb, M, Nn, K, ldf, co, act);
  };

  // ---- weight conversion (bf16, K padded to 32) ----
  bf16* fcW0 = convW(IDX_FC + 0, 512, 1027, 1056);
  bf16* fcW1 = convW(IDX_FC + 2, 512, 512, 512);
  bf16* fcW2 = convW(IDX_FC + 4, 512, 512, 512);
  bf16 *fdW0[4], *fdW1[4], *fdW2[4], *fdM2[4][3], *mseM2[4][3];
  for (int blk = 0; blk < 4; ++blk) {
    int base = IDX_FD + blk * 12;
    fdW0[blk] = convW(base + 0, 512, 1030, 1056);
    fdW1[blk] = convW(base + 2, 256, 512, 512);
    fdW2[blk] = convW(base + 4, 64, 256, 256);
    for (int l = 0; l < 3; ++l) fdM2[blk][l] = convW(base + 6 + l * 2, 64, 64, 64);
  }
  for (int blk = 0; blk < 4; ++blk)
    for (int l = 0; l < 3; ++l) mseM2[blk][l] = convW(IDX_MSE + blk * 12 + 6 + l * 2, 64, 64, 64);
  bf16* fpW1 = convW(IDX_FP + 0, 256, 512, 512);
  bf16* fpW2 = convW(IDX_FP + 2, 128, 256, 256);
  bf16* fpW3 = convW(IDX_FP + 4, 64, 128, 128);

  // ---- transposes + KNN ----
  int t3 = B * N * 3;
  transpose3_kernel<<<ceilDiv(t3, 256), 256, 0, stream>>>(P[0], x1, N);
  transpose3_kernel<<<ceilDiv(t3, 256), 256, 0, stream>>>(P[1], x2, N);
  transpose3_kernel<<<ceilDiv(t3, 256), 256, 0, stream>>>(P[2], f1t, N);
  transpose3_kernel<<<ceilDiv(t3, 256), 256, 0, stream>>>(P[3], f2t, N);
  {
    dim3 g(ceilDiv(N, 256), B);
    knn_kernel<<<g, 256, 0, stream>>>(x1, x1, knn1, 32, N);
    knn_kernel<<<g, 256, 0, stream>>>(x2, x2, knn2, 32, N);
    knn_kernel<<<g, 256, 0, stream>>>(x1, x2, knnc, 8, N);
  }

  // ---- mse on both clouds ----
  for (int cl = 0; cl < 2; ++cl) {
    const float* xt = cl ? x2 : x1;
    const float* ft = cl ? f2t : f1t;
    const int* kn = cl ? knn2 : knn1;
    float* pf = cl ? pf2 : pf1;
    float* pfm = cl ? pf2m : pf1m;
    for (int blk = 0; blk < 4; ++blk) {
      int k = SA[blk];
      int base = IDX_MSE + blk * 12;
      (void)hipMemsetAsync(pre64, 0, (size_t)B * N * 64 * 4, stream);
      int tot = B * N * k;
      mse_block_kernel<<<ceilDiv(tot, 128), 128, 0, stream>>>(
          xt, ft, kn, k, P[base + 0], P[base + 1], P[base + 2], P[base + 3], P[base + 4],
          P[base + 5], (unsigned int*)pre64, N);
      f32_to_bf16_kernel<<<ceilDiv(B * N * 64, 256), 256, 0, stream>>>(pre64, Xm64, B * N * 64);
      gemm(Xm64, mseM2[blk][0], P[base + 7], nullptr, 0, 0, Ym64a, B * N, 64, 64, 1);
      gemm(Ym64a, mseM2[blk][1], P[base + 9], nullptr, 0, 0, Ym64b, B * N, 64, 64, 1);
      gemm(Ym64b, mseM2[blk][2], P[base + 11], pf, 256, blk * 64, nullptr, B * N, 64, 64, 1);
    }
    colmax_kernel<<<ceilDiv(B * 256, 256), 256, 0, stream>>>(pf, pfm, N, 256);
  }

  // ---- fc (flow correlation), chunked 8192 rows ----
  for (int ch = 0; ch < 2; ++ch) {
    int rb = ch * 8192, rows = 8192, pBase = ch * 1024;
    gather_fc_kernel<<<ceilDiv(rows * 1056, 256), 256, 0, stream>>>(pf1, pf1m, pf2, pf2m, x1, x2,
                                                                    knnc, Xbig, N, rb, rows);
    gemm(Xbig, fcW0, P[IDX_FC + 1], nullptr, 0, 0, Yba, rows, 512, 1056, 2);
    gemm(Yba, fcW1, P[IDX_FC + 3], nullptr, 0, 0, Ybb, rows, 512, 512, 2);
    gemm(Ybb, fcW2, P[IDX_FC + 5], hF32, 512, 0, nullptr, rows, 512, 512, 2);
    wn_kernel<<<ceilDiv(rows, 128), 128, 0, stream>>>(x1, x2, knnc, 8, P[IDX_WN1 + 0],
                                                      P[IDX_WN1 + 1], P[IDX_WN1 + 2],
                                                      P[IDX_WN1 + 3], P[IDX_WN1 + 4],
                                                      P[IDX_WN1 + 5], wF32, N, rb, rows);
    p2p_sum_kernel<<<ceilDiv(1024 * 512, 256), 256, 0, stream>>>(wF32, hF32, p2p, pBase, 1024);
  }
  for (int ch = 0; ch < 2; ++ch) {
    int rb = ch * 8192, rows = 8192, pBase = ch * 1024;
    wn_kernel<<<ceilDiv(rows, 128), 128, 0, stream>>>(x1, x1, knn1, 32, P[IDX_WN2 + 0],
                                                      P[IDX_WN2 + 1], P[IDX_WN2 + 2],
                                                      P[IDX_WN2 + 3], P[IDX_WN2 + 4],
                                                      P[IDX_WN2 + 5], wF32, N, rb, rows);
    cor_kernel<<<ceilDiv(1024 * 512, 256), 256, 0, stream>>>(wF32, p2p, knn1, cor, N, pBase, 1024);
  }

  // ---- emb + fd_mse ----
  emb_kernel<<<ceilDiv(B * N * 1027, 256), 256, 0, stream>>>(f1t, pf1, pf1m, cor, emb, N);
  for (int blk = 0; blk < 4; ++blk) {
    int k = SA[blk];
    int base = IDX_FD + blk * 12;
    (void)hipMemsetAsync(pre64, 0, (size_t)B * N * 64 * 4, stream);
    int total = B * N * k;
    for (int rb = 0; rb < total; rb += 8192) {
      int rows = 8192;
      gather_fd_kernel<<<ceilDiv(rows * 1056, 256), 256, 0, stream>>>(emb, x1, knn1, k, Xbig, N,
                                                                      rb, rows);
      gemm(Xbig, fdW0[blk], P[base + 1], nullptr, 0, 0, Yba, rows, 512, 1056, 1);
      gemm(Yba, fdW1[blk], P[base + 3], nullptr, 0, 0, Ybb, rows, 256, 512, 1);
      gemm(Ybb, fdW2[blk], P[base + 5], Y3f, 64, 0, nullptr, rows, 64, 256, 1);
      fdmax_kernel<<<ceilDiv((rows / k) * 64, 256), 256, 0, stream>>>(Y3f, (unsigned int*)pre64, k,
                                                                      rb, rows);
    }
    f32_to_bf16_kernel<<<ceilDiv(B * N * 64, 256), 256, 0, stream>>>(pre64, Xm64, B * N * 64);
    gemm(Xm64, fdM2[blk][0], P[base + 7], nullptr, 0, 0, Ym64a, B * N, 64, 64, 1);
    gemm(Ym64a, fdM2[blk][1], P[base + 9], nullptr, 0, 0, Ym64b, B * N, 64, 64, 1);
    gemm(Ym64b, fdM2[blk][2], P[base + 11], prop, 256, blk * 64, nullptr, B * N, 64, 64, 1);
  }
  colmax_kernel<<<ceilDiv(B * 256, 256), 256, 0, stream>>>(prop, propm, N, 256);

  // ---- fp head ----
  gather_fp_kernel<<<ceilDiv(B * N * 512, 256), 256, 0, stream>>>(prop, propm, Xfp, N);
  gemm(Xfp, fpW1, P[IDX_FP + 1], nullptr, 0, 0, fpYa, B * N, 256, 512, 1);
  gemm(fpYa, fpW2, P[IDX_FP + 3], nullptr, 0, 0, fpYb, B * N, 128, 256, 1);
  gemm(fpYb, fpW3, P[IDX_FP + 5], fpY3, 64, 0, nullptr, B * N, 64, 128, 1);
  fp_out_kernel<<<ceilDiv(B * N, 128), 128, 0, stream>>>(fpY3, P[IDX_FP + 6], P[IDX_FP + 7], outp,
                                                         N);

  // ---- rigid refinement (_sfr) ----
  rigid_kernel<<<B, 256, 0, stream>>>(P[0], outp, nullptr, trans1, N);
  (void)hipMemsetAsync(countf, 0, B * 4, stream);
  mask_kernel<<<ceilDiv(B * N, 256), 256, 0, stream>>>(P[0], P[2], P[4], trans1, mask_out, countf,
                                                       N);
  rigid_kernel<<<B, 256, 0, stream>>>(P[0], outp, mask_out, trans2, N);
  final_kernel<<<ceilDiv(B * N, 256), 256, 0, stream>>>(P[0], outp, trans1, trans2, mask_out,
                                                        countf, sf_agg, pre_trans, N);
}
